// Attention_15015205667492
// MI455X (gfx1250) — matrix operand order repaired and checked by reading, not compiled, not run on hardware
//
#include <hip/hip_runtime.h>

#define B_ 2
#define S_ 2048
#define D_ 2048
#define H_ 32
#define KV_ 8
#define HD_ 64
#define M_ (B_*S_)   // 4096 rows in all big GEMMs

typedef __bf16 v16bf __attribute__((ext_vector_type(16)));
typedef float  v8f   __attribute__((ext_vector_type(8)));

union Frag { v16bf bf; int4 i4[2]; };

static __device__ __forceinline__ unsigned int f2bf(float f) {
  unsigned int u = __float_as_uint(f);
  u += 0x7fffu + ((u >> 16) & 1u);          // round-to-nearest-even
  return u >> 16;
}

static __device__ __forceinline__ v8f wmma_bf16(v16bf a, v16bf b, v8f c) {
  return __builtin_amdgcn_wmma_f32_16x16x32_bf16(false, a, false, b, (short)0, c,
                                                 false, false);
}

// ---------------- elementwise f32 -> bf16 ----------------
__global__ void k_cvt(const float* __restrict__ in, unsigned short* __restrict__ out) {
  int i = blockIdx.x * 256 + threadIdx.x;
  out[i] = (unsigned short)f2bf(in[i]);
}

// -------- tiled transpose + convert: f32 KxN -> bf16 NxK --------
__global__ void k_transpose_cvt(const float* __restrict__ in,
                                unsigned short* __restrict__ out, int K, int N) {
  __shared__ float tile[32][33];
  const int n0 = blockIdx.x * 32, k0 = blockIdx.y * 32;
  const int tx = threadIdx.x, ty = threadIdx.y;   // (32, 8)
  #pragma unroll
  for (int i = 0; i < 32; i += 8)
    tile[ty + i][tx] = in[(size_t)(k0 + ty + i) * N + (n0 + tx)];
  __syncthreads();
  #pragma unroll
  for (int i = 0; i < 32; i += 8)
    out[(size_t)(n0 + ty + i) * K + (k0 + tx)] = (unsigned short)f2bf(tile[tx][ty + i]);
}

// ===== shared 32x64-wave-tile GEMM core: 12 loads + 8 WMMA per K step =====
// A: M x K bf16 row-major; Bt: N x K bf16 row-major (B transposed).
// No explicit prefetch: the compiler software-pipelines these loads with
// partial s_wait_loadcnt (verified in disasm); prefetch hints only add
// register pressure and break the pipeline.
#define GEMM_CORE(ap, bp, K, acc)                                              \
  for (int kk = 0; kk < (K); kk += 32) {                                       \
    Frag a[2], b[4];                                                           \
    _Pragma("unroll")                                                          \
    for (int i = 0; i < 2; ++i) {                                              \
      const int4* pa = (const int4*)((ap)[i] + kk + half * 8);                 \
      a[i].i4[0] = pa[0]; a[i].i4[1] = pa[2];                                  \
    }                                                                          \
    _Pragma("unroll")                                                          \
    for (int j = 0; j < 4; ++j) {                                              \
      const int4* pb = (const int4*)((bp)[j] + kk + half * 8);                 \
      b[j].i4[0] = pb[0]; b[j].i4[1] = pb[2];                                  \
    }                                                                          \
    _Pragma("unroll")                                                          \
    for (int i = 0; i < 2; ++i)                                                \
      _Pragma("unroll")                                                        \
      for (int j = 0; j < 4; ++j)                                              \
        (acc)[i][j] = wmma_bf16(a[i].bf, b[j].bf, (acc)[i][j]);                \
  }

// -------- plain GEMM (32x64 wave tile), f32 out: output projection --------
__global__ __launch_bounds__(256) void k_gemm_out(
    const unsigned short* __restrict__ A, const unsigned short* __restrict__ Bt,
    float* __restrict__ C, int N, int K) {
  const int wid  = threadIdx.x >> 5;
  const int lane = threadIdx.x & 31;
  const int l15  = lane & 15;
  const int half = lane >> 4;
  const int nw   = N >> 6;
  const int w    = blockIdx.x * 8 + wid;
  const int tm   = (w / nw) << 5;
  const int tn   = (w % nw) << 6;

  v8f acc[2][4] = {};
  const unsigned short* ap[2] = { A + (size_t)(tm + l15) * K,
                                  A + (size_t)(tm + 16 + l15) * K };
  const unsigned short* bp[4] = { Bt + (size_t)(tn + l15) * K,
                                  Bt + (size_t)(tn + 16 + l15) * K,
                                  Bt + (size_t)(tn + 32 + l15) * K,
                                  Bt + (size_t)(tn + 48 + l15) * K };
  GEMM_CORE(ap, bp, K, acc)

  #pragma unroll
  for (int i = 0; i < 2; ++i)
    #pragma unroll
    for (int r = 0; r < 8; ++r) {
      const int row = tm + i * 16 + half * 8 + r;
      #pragma unroll
      for (int j = 0; j < 4; ++j)
        C[(size_t)row * N + tn + j * 16 + l15] = acc[i][j][r];
    }
}

// -------- projection GEMM with fused RoPE + bf16 store --------
// wave tile 32x64, head-aligned (N-dim = h*64 + hd); rotate-half partner of
// accumulator acc[i][j] is acc[i][j^2] (hd +/- 32), so RoPE is done in-register.
__global__ __launch_bounds__(256) void k_gemm_rope(
    const unsigned short* __restrict__ A,   // M x K bf16
    const unsigned short* __restrict__ Bt,  // N x K bf16
    const float* __restrict__ cs, const float* __restrict__ sn,  // (S, 64)
    unsigned short* __restrict__ Out,       // (b,s,NH,64) bf16, RoPE applied
    int N, int K) {
  const int wid  = threadIdx.x >> 5;
  const int lane = threadIdx.x & 31;
  const int l15  = lane & 15;
  const int half = lane >> 4;
  const int nw   = N >> 6;
  const int w    = blockIdx.x * 8 + wid;
  const int tm   = (w / nw) << 5;
  const int tn   = (w % nw) << 6;           // one full head

  v8f acc[2][4] = {};
  const unsigned short* ap[2] = { A + (size_t)(tm + l15) * K,
                                  A + (size_t)(tm + 16 + l15) * K };
  const unsigned short* bp[4] = { Bt + (size_t)(tn + l15) * K,
                                  Bt + (size_t)(tn + 16 + l15) * K,
                                  Bt + (size_t)(tn + 32 + l15) * K,
                                  Bt + (size_t)(tn + 48 + l15) * K };
  GEMM_CORE(ap, bp, K, acc)

  // epilogue: RoPE in-register, write bf16
  #pragma unroll
  for (int i = 0; i < 2; ++i) {
    #pragma unroll
    for (int r = 0; r < 8; ++r) {
      const int row = tm + i * 16 + half * 8 + r;     // b*S + s
      const int si  = row & (S_ - 1);
      #pragma unroll
      for (int j = 0; j < 4; ++j) {
        const int hd = j * 16 + l15;
        const float v       = acc[i][j][r];
        const float partner = acc[i][j ^ 2][r];
        const float rot     = (j < 2) ? -partner : partner;
        const float c = cs[(si << 6) + hd];
        const float s = sn[(si << 6) + hd];
        Out[(size_t)row * N + tn + hd] = (unsigned short)f2bf(fmaf(v, c, rot * s));
      }
    }
  }
}

// -------- V projection GEMM with fused transpose to (b,kv,hd,s) bf16 --------
// wave tile 32x64; per (i,j) the 8 accumulator rows are 8 consecutive s values
// for one hd column -> pack into a single 16B store (coalesced transpose).
__global__ __launch_bounds__(256) void k_gemm_vt(
    const unsigned short* __restrict__ A,   // M x K bf16
    const unsigned short* __restrict__ Bt,  // N x K bf16 (N = KV*HD = 512)
    unsigned short* __restrict__ Vt,        // (b,kv,hd,s) bf16
    int N, int K) {
  const int wid  = threadIdx.x >> 5;
  const int lane = threadIdx.x & 31;
  const int l15  = lane & 15;
  const int half = lane >> 4;
  const int nw   = N >> 6;
  const int w    = blockIdx.x * 8 + wid;
  const int tm   = (w / nw) << 5;
  const int tn   = (w % nw) << 6;

  v8f acc[2][4] = {};
  const unsigned short* ap[2] = { A + (size_t)(tm + l15) * K,
                                  A + (size_t)(tm + 16 + l15) * K };
  const unsigned short* bp[4] = { Bt + (size_t)(tn + l15) * K,
                                  Bt + (size_t)(tn + 16 + l15) * K,
                                  Bt + (size_t)(tn + 32 + l15) * K,
                                  Bt + (size_t)(tn + 48 + l15) * K };
  GEMM_CORE(ap, bp, K, acc)

  const int bi     = tm / S_;
  const int sbase0 = tm & (S_ - 1);
  #pragma unroll
  for (int i = 0; i < 2; ++i) {
    const int sb = sbase0 + i * 16 + half * 8;      // 8-aligned
    #pragma unroll
    for (int j = 0; j < 4; ++j) {
      const int col = tn + j * 16 + l15;            // kvh*64 + hd
      const int kvh = col >> 6;
      const int hd  = col & 63;
      int4 pk;
      pk.x = (int)(f2bf(acc[i][j][0]) | (f2bf(acc[i][j][1]) << 16));
      pk.y = (int)(f2bf(acc[i][j][2]) | (f2bf(acc[i][j][3]) << 16));
      pk.z = (int)(f2bf(acc[i][j][4]) | (f2bf(acc[i][j][5]) << 16));
      pk.w = (int)(f2bf(acc[i][j][6]) | (f2bf(acc[i][j][7]) << 16));
      *(int4*)(Vt + (((size_t)(bi * KV_ + kvh) * HD_ + hd) << 11) + sb) = pk;
    }
  }
}

// -------- flash attention: one wave per (b, h, 16-query tile) --------
__global__ __launch_bounds__(256) void k_flash(
    const unsigned short* __restrict__ Qb,   // (b,s,H,HD) bf16, RoPE'd
    const unsigned short* __restrict__ Kb,   // (b,s,KV,HD) bf16, RoPE'd
    const unsigned short* __restrict__ Vt,   // (b,KV,HD,s) bf16
    unsigned short* __restrict__ Ob) {       // (b,s,H,HD) bf16
  __shared__ unsigned short ldsP[8][16 * 32];
  const int wid  = threadIdx.x >> 5;
  const int lane = threadIdx.x & 31;
  const int l15  = lane & 15;
  const int half = lane >> 4;
  const int QT   = S_ / 16;
  const int w    = blockIdx.x * 8 + wid;
  const int qt   = w % QT;
  const int h    = (w / QT) % H_;
  const int bi   = w / (QT * H_);
  const int kvh  = h >> 2;                    // REP = 4
  const int qbase = qt * 16;

  // Q fragments (A operand, 16x32 each; hd 0..31 and 32..63)
  const unsigned short* qp =
      Qb + (((size_t)(bi * S_ + qbase + l15) * H_ + h) << 6);
  Frag qa0, qa1;
  { const int4* q4  = (const int4*)(qp + half * 8);
    qa0.i4[0] = q4[0];  qa0.i4[1] = q4[2];
    const int4* q4b = (const int4*)(qp + 32 + half * 8);
    qa1.i4[0] = q4b[0]; qa1.i4[1] = q4b[2]; }

  float m[8], l[8];
  v8f o[4] = {};
  #pragma unroll
  for (int r = 0; r < 8; ++r) { m[r] = -1e30f; l[r] = 0.f; }

  const int nch = (qbase + 16 + 31) >> 5;     // 32-key chunks touching causal region
  for (int ch = 0; ch < nch; ++ch) {
    const int jb = ch * 32;

    // prefetch next chunk's K/V rows; consumed only after the long
    // WMMA -> softmax -> WMMA chain of this chunk (speculative, ISA-dropped
    // if useless), so here prefetch genuinely runs ahead of use.
    { const int jn = jb + 32;
      __builtin_prefetch(Kb + (((size_t)(bi * S_ + jn + l15) * KV_ + kvh) << 6), 0, 3);
      __builtin_prefetch(Kb + (((size_t)(bi * S_ + jn + 16 + l15) * KV_ + kvh) << 6), 0, 3);
      __builtin_prefetch(Vt + (((size_t)(bi * KV_ + kvh) * HD_ + l15) << 11) + jn, 0, 3);
      __builtin_prefetch(Vt + (((size_t)(bi * KV_ + kvh) * HD_ + 32 + l15) << 11) + jn, 0, 3); }

    // scores: S[16q x 32k] = Q(16x64) . K^T, two 16x16 WMMA tiles over key cols
    v8f s0acc = {}, s1acc = {};
    { const unsigned short* kp =
          Kb + (((size_t)(bi * S_ + jb + l15) * KV_ + kvh) << 6);
      Frag kb0, kb1;
      const int4* k4  = (const int4*)(kp + half * 8);
      kb0.i4[0] = k4[0];  kb0.i4[1] = k4[2];
      const int4* k4b = (const int4*)(kp + 32 + half * 8);
      kb1.i4[0] = k4b[0]; kb1.i4[1] = k4b[2];
      s0acc = wmma_bf16(qa0.bf, kb0.bf, s0acc);
      s0acc = wmma_bf16(qa1.bf, kb1.bf, s0acc); }
    { const unsigned short* kp =
          Kb + (((size_t)(bi * S_ + jb + 16 + l15) * KV_ + kvh) << 6);
      Frag kb0, kb1;
      const int4* k4  = (const int4*)(kp + half * 8);
      kb0.i4[0] = k4[0];  kb0.i4[1] = k4[2];
      const int4* k4b = (const int4*)(kp + 32 + half * 8);
      kb1.i4[0] = k4b[0]; kb1.i4[1] = k4b[2];
      s1acc = wmma_bf16(qa0.bf, kb0.bf, s1acc);
      s1acc = wmma_bf16(qa1.bf, kb1.bf, s1acc); }

    // online softmax across this 32-key chunk (per-row over 16 lanes)
    #pragma unroll
    for (int r = 0; r < 8; ++r) {
      const int qi = qbase + r + half * 8;
      float s0 = s0acc[r] * 0.125f;           // 1/sqrt(64)
      float s1 = s1acc[r] * 0.125f;
      if (jb + l15 > qi)      s0 = -1e30f;    // causal mask (arith select, EXEC intact)
      if (jb + 16 + l15 > qi) s1 = -1e30f;
      float mx = fmaxf(s0, s1);
      mx = fmaxf(mx, __shfl_xor(mx, 8, 16));
      mx = fmaxf(mx, __shfl_xor(mx, 4, 16));
      mx = fmaxf(mx, __shfl_xor(mx, 2, 16));
      mx = fmaxf(mx, __shfl_xor(mx, 1, 16));
      const float mn   = fmaxf(m[r], mx);
      const float corr = __expf(m[r] - mn);
      const float p0   = __expf(s0 - mn);
      const float p1   = __expf(s1 - mn);
      float rs = p0 + p1;
      rs += __shfl_xor(rs, 8, 16);
      rs += __shfl_xor(rs, 4, 16);
      rs += __shfl_xor(rs, 2, 16);
      rs += __shfl_xor(rs, 1, 16);
      l[r] = l[r] * corr + rs;
      m[r] = mn;
      #pragma unroll
      for (int nt = 0; nt < 4; ++nt) o[nt][r] *= corr;
      const int prow = r + half * 8;          // C-layout row -> LDS row-major P
      ldsP[wid][prow * 32 + l15]      = (unsigned short)f2bf(p0);
      ldsP[wid][prow * 32 + 16 + l15] = (unsigned short)f2bf(p1);
    }

    // reload P as A fragment (16x32) from LDS
    Frag pa;
    { const int4* p4 = (const int4*)(&ldsP[wid][0] + l15 * 32 + half * 8);
      pa.i4[0] = p4[0]; pa.i4[1] = p4[2]; }

    // O(16x64) += P(16x32) . V(32x64) : 4 WMMA over hd tiles
    #pragma unroll
    for (int nt = 0; nt < 4; ++nt) {
      const unsigned short* vp =
          Vt + (((size_t)(bi * KV_ + kvh) * HD_ + nt * 16 + l15) << 11) + jb;
      Frag vb;
      const int4* v4 = (const int4*)(vp + half * 8);
      vb.i4[0] = v4[0]; vb.i4[1] = v4[2];
      o[nt] = wmma_bf16(pa.bf, vb.bf, o[nt]);
    }
  }

  // normalize + store bf16
  #pragma unroll
  for (int r = 0; r < 8; ++r) {
    const float inv = 1.0f / l[r];
    const int qi = qbase + r + half * 8;
    const size_t base = ((size_t)(bi * S_ + qi) * H_ + h) << 6;
    #pragma unroll
    for (int nt = 0; nt < 4; ++nt)
      Ob[base + nt * 16 + l15] = (unsigned short)f2bf(o[nt][r] * inv);
  }
}

extern "C" void kernel_launch(void* const* d_in, const int* in_sizes, int n_in,
                              void* d_out, int out_size, void* d_ws, size_t ws_size,
                              hipStream_t stream) {
  const float* x    = (const float*)d_in[0];
  const float* cosp = (const float*)d_in[1];
  const float* sinp = (const float*)d_in[2];
  const float* wq   = (const float*)d_in[3];
  const float* wk   = (const float*)d_in[4];
  const float* wv   = (const float*)d_in[5];
  const float* wo   = (const float*)d_in[6];
  float* out = (float*)d_out;

  char* ws = (char*)d_ws;
  size_t off = 0;
  auto alloc = [&](size_t bytes) {
    char* p = ws + off;
    off += (bytes + 255) & ~(size_t)255;
    return p;
  };
  unsigned short* xb  = (unsigned short*)alloc((size_t)M_ * D_ * 2);        // x bf16
  unsigned short* wqT = (unsigned short*)alloc((size_t)2048 * 2048 * 2);    // N x K
  unsigned short* wkT = (unsigned short*)alloc((size_t)512  * 2048 * 2);
  unsigned short* wvT = (unsigned short*)alloc((size_t)512  * 2048 * 2);
  unsigned short* woT = (unsigned short*)alloc((size_t)2048 * 2048 * 2);
  unsigned short* Qb  = (unsigned short*)alloc((size_t)M_ * 2048 * 2);      // RoPE'd Q bf16
  unsigned short* Kb  = (unsigned short*)alloc((size_t)M_ * 512 * 2);       // RoPE'd K bf16
  unsigned short* Vt  = (unsigned short*)alloc((size_t)M_ * 512 * 2);       // V^T bf16
  unsigned short* Ob  = (unsigned short*)alloc((size_t)M_ * 2048 * 2);      // attn out bf16

  // 1) precision conversion + weight transposes
  k_cvt<<<(M_ * D_) / 256, 256, 0, stream>>>(x, xb);
  k_transpose_cvt<<<dim3(2048 / 32, 2048 / 32), dim3(32, 8), 0, stream>>>(wq, wqT, 2048, 2048);
  k_transpose_cvt<<<dim3(512  / 32, 2048 / 32), dim3(32, 8), 0, stream>>>(wk, wkT, 2048, 512);
  k_transpose_cvt<<<dim3(512  / 32, 2048 / 32), dim3(32, 8), 0, stream>>>(wv, wvT, 2048, 512);
  k_transpose_cvt<<<dim3(2048 / 32, 2048 / 32), dim3(32, 8), 0, stream>>>(wo, woT, 2048, 2048);

  // 2) fused projections: GEMM + RoPE + bf16 (Q,K); GEMM + transpose + bf16 (V)
  k_gemm_rope<<<(M_ / 32) * (2048 / 64) / 8, 256, 0, stream>>>(xb, wqT, cosp, sinp, Qb, 2048, 2048);
  k_gemm_rope<<<(M_ / 32) * (512  / 64) / 8, 256, 0, stream>>>(xb, wkT, cosp, sinp, Kb, 512,  2048);
  k_gemm_vt  <<<(M_ / 32) * (512  / 64) / 8, 256, 0, stream>>>(xb, wvT, Vt, 512, 2048);

  // 3) flash attention (one wave per (b,h,16-query tile))
  k_flash<<<(B_ * H_ * (S_ / 16)) / 8, 256, 0, stream>>>(Qb, Kb, Vt, Ob);

  // 4) output projection -> d_out (f32), 32x64 wave tile
  k_gemm_out<<<(M_ / 32) * (2048 / 64) / 8, 256, 0, stream>>>(Ob, woT, out, 2048, 2048);
}
